// OrientedRCNNHead_29137058136625
// MI455X (gfx1250) — matrix operand order repaired and checked by reading, not compiled, not run on hardware
//
#include <hip/hip_runtime.h>
#include <hip/hip_bf16.h>

// ---------- types ----------
typedef __bf16 bf16;
typedef bf16 bf16x16 __attribute__((ext_vector_type(16)));
typedef float f32x8 __attribute__((ext_vector_type(8)));
typedef unsigned int u32x4 __attribute__((ext_vector_type(4)));
typedef unsigned int u32x2 __attribute__((ext_vector_type(2)));

union FragU { bf16x16 v; u32x4 q[2]; };

// native convert: clang emits v_cvt_pk_bf16_f32 (RNE)
__device__ __forceinline__ unsigned short f2bf(float f) {
    union { bf16 h; unsigned short u; } c;
    c.h = (bf16)f;
    return c.u;
}
__device__ __forceinline__ float bf2f(unsigned short h) {
    union { unsigned int u; float f; } c; c.u = ((unsigned int)h) << 16;
    return c.f;
}

// ---------- CDNA5 async global->LDS (ASYNCcnt-tracked) ----------
// INST_OFFSET is applied to BOTH the LDS and global address (ISA 15.18.3 op 98).
__device__ __forceinline__ void async_b128(unsigned lds_off, const void* g) {
    asm volatile("global_load_async_to_lds_b128 %0, %1, off"
                 :: "v"(lds_off), "v"(g) : "memory");
}
__device__ __forceinline__ void async_b128_o16(unsigned lds_off, const void* g) {
    asm volatile("global_load_async_to_lds_b128 %0, %1, off offset:16"
                 :: "v"(lds_off), "v"(g) : "memory");
}
__device__ __forceinline__ void wait_async0() {
    asm volatile("s_wait_asynccnt 0x0" ::: "memory");
}
__device__ __forceinline__ unsigned lds_addr(const void* p) {
    return (unsigned)(unsigned long long)p;
}

// ---------- elementwise fp32 -> bf16 (features): makes A L2-resident for GEMM1 ----------
__global__ __launch_bounds__(256) void convert_bf16(const float* __restrict__ in,
                                                    unsigned short* __restrict__ out,
                                                    long n4) {   // n4 = count/4
    long i = (long)blockIdx.x * blockDim.x + threadIdx.x;
    const long stride = (long)gridDim.x * blockDim.x;
    for (; i < n4; i += stride) {
        const u32x4 v = ((const u32x4*)in)[i];
        const float* f = (const float*)&v;
        bf16 t[4];
#pragma unroll
        for (int j = 0; j < 4; ++j) t[j] = (bf16)f[j];
        *(u32x2*)&out[i * 4] = *(const u32x2*)&t[0];
    }
}

// ---------- W transpose + fp32->bf16 convert: out[n*K+k] = bf16(in[k*N+n]) ----------
__global__ __launch_bounds__(256) void transpose_bf16(const float* __restrict__ in,
                                                      unsigned short* __restrict__ out,
                                                      int K, int N) {
    __shared__ unsigned short tile[32][33];
    const int k0 = blockIdx.x * 32, n0 = blockIdx.y * 32;
    const int tx = threadIdx.x, ty = threadIdx.y;     // 32 x 8
#pragma unroll
    for (int i = 0; i < 32; i += 8) {
        tile[ty + i][tx] = f2bf(in[(long)(k0 + ty + i) * N + (n0 + tx)]);
    }
    __syncthreads();
#pragma unroll
    for (int i = 0; i < 32; i += 8) {
        out[(long)(n0 + ty + i) * K + (k0 + tx)] = tile[tx][ty + i];
    }
}

// ---------- WMMA GEMM: C[M,N] = relu(A[M,K] @ Bt[N,K]^T + bias), all bf16 ----------
// Block tile 128x128, BK=32, 256 threads = 8 waves (2 M x 4 N), wave tile 64x32.
// Double-buffered LDS; BOTH operand tiles stream via global_load_async_to_lds_b128,
// so the inner loop is: s_wait_asynccnt -> barrier -> issue next DMAs -> ds_load frags -> WMMA.
#define LDS_STRIDE 40   // 32 + 8 pad (80B row stride -> conflict-free b128 frag loads)

__global__ __launch_bounds__(256) void gemm_relu_wmma(const unsigned short* __restrict__ A,
                                                      const unsigned short* __restrict__ Bt,
                                                      const float* __restrict__ bias,
                                                      unsigned short* __restrict__ C,
                                                      int M, int N, int K) {
    __shared__ unsigned short As[2][128 * LDS_STRIDE];
    __shared__ unsigned short Bs[2][128 * LDS_STRIDE];

    const int tid  = threadIdx.x;
    const int lane = tid & 31;
    const int wid  = tid >> 5;
    const int wm   = wid & 1;          // 0..1  -> 64-row half
    const int wn   = wid >> 1;         // 0..3  -> 32-col quarter
    const int mTile = blockIdx.y * 128;
    const int nTile = blockIdx.x * 128;

    const int loadRow = tid >> 1;          // 0..127
    const int loadK   = (tid & 1) * 16;    // 0 or 16 (16 elems each)
    const int ldsBase = loadRow * LDS_STRIDE + loadK;

    f32x8 acc[4][2] = {};

    auto asyncTile = [&](int kt, int buf) {
        const unsigned short* asrc = A  + (long)(mTile + loadRow) * K + (long)kt * 32 + loadK;
        const unsigned short* bsrc = Bt + (long)(nTile + loadRow) * K + (long)kt * 32 + loadK;
        const unsigned adst = lds_addr(&As[buf][ldsBase]);
        const unsigned bdst = lds_addr(&Bs[buf][ldsBase]);
        async_b128(adst, asrc);
        async_b128_o16(adst, asrc);   // +16B in both LDS and global
        async_b128(bdst, bsrc);
        async_b128_o16(bdst, bsrc);
    };

    const int nk = K / 32;
    asyncTile(0, 0);                  // prologue: tile 0 in flight

    const int mBase = wm * 64;
    const int nBase = wn * 32;
    const int fRow  = lane & 15;
    const int aK    = (lane >> 4) * 8;    // A: chunks at aK and aK+16
    const int bK    = (lane >> 4) * 8;    // B: chunks at bK and bK+16

    for (int kt = 0; kt < nk; ++kt) {
        const int cur = kt & 1, nxt = cur ^ 1;

        wait_async0();                 // tile kt fully landed in LDS
        __syncthreads();

        if (kt + 1 < nk) asyncTile(kt + 1, nxt);   // DMA overlaps WMMA phase

        FragU a[4], b[2];
#pragma unroll
        for (int mi = 0; mi < 4; ++mi) {
            const int r = (mBase + mi * 16 + fRow) * LDS_STRIDE;
            a[mi].q[0] = *(const u32x4*)&As[cur][r + aK];
            a[mi].q[1] = *(const u32x4*)&As[cur][r + aK + 16];
        }
#pragma unroll
        for (int ni = 0; ni < 2; ++ni) {
            const int r = (nBase + ni * 16 + fRow) * LDS_STRIDE;
            b[ni].q[0] = *(const u32x4*)&Bs[cur][r + bK];
            b[ni].q[1] = *(const u32x4*)&Bs[cur][r + bK + 16];
        }
#pragma unroll
        for (int mi = 0; mi < 4; ++mi)
#pragma unroll
            for (int ni = 0; ni < 2; ++ni)
                acc[mi][ni] = __builtin_amdgcn_wmma_f32_16x16x32_bf16(
                    false, a[mi].v, false, b[ni].v, (short)0, acc[mi][ni], false, false);
    }

    // C/D layout: VGPR v, lane L -> M = v + 8*(L>=16), N = L%16
    const int colLane = lane & 15;
    const int rowSel  = (lane >> 4) * 8;
#pragma unroll
    for (int mi = 0; mi < 4; ++mi)
#pragma unroll
        for (int ni = 0; ni < 2; ++ni) {
            const int col = nTile + nBase + ni * 16 + colLane;
            const float bv = bias[col];
#pragma unroll
            for (int v = 0; v < 8; ++v) {
                const int row = mTile + mBase + mi * 16 + rowSel + v;
                float val = acc[mi][ni][v] + bv;
                val = val > 0.f ? val : 0.f;
                C[(long)row * N + col] = f2bf(val);
            }
        }
}

// ---------- head: cls/reg projections + normalize + box decode + rois ----------
__global__ __launch_bounds__(256) void head_kernel(const unsigned short* __restrict__ h2,
                                                   const float* __restrict__ Wc,
                                                   const float* __restrict__ bc,
                                                   const float* __restrict__ Wr,
                                                   const float* __restrict__ br,
                                                   const float* __restrict__ boxes,
                                                   const float* __restrict__ strides,
                                                   float* __restrict__ outCls,
                                                   float* __restrict__ outDec,
                                                   float* __restrict__ outRois,
                                                   int rowsTotal) {
    __shared__ unsigned short Wl[1024 * 16];  // [k][o] packed bf16, o: 0..10 cls, 11..15 reg
    __shared__ float hs[1024];
    __shared__ float rbuf[5];

    const int tid = threadIdx.x;
    for (int i = tid; i < 1024 * 16; i += 256) {
        const int k = i >> 4, o = i & 15;
        const float w = (o < 11) ? Wc[(long)k * 11 + o] : Wr[(long)k * 5 + (o - 11)];
        Wl[i] = f2bf(w);
    }
    __syncthreads();

    const int o = tid >> 4, g = tid & 15;
    const float STD[5] = {0.1f, 0.1f, 0.2f, 0.2f, 0.1f};
    const int rowsPerBlk = rowsTotal / gridDim.x;

    for (int r = 0; r < rowsPerBlk; ++r) {
        const int row = blockIdx.x * rowsPerBlk + r;
        for (int i = tid; i < 1024; i += 256) hs[i] = bf2f(h2[(long)row * 1024 + i]);
        __syncthreads();

        float acc = 0.f;
        const int k0 = g * 64;
#pragma unroll 8
        for (int kk = 0; kk < 64; ++kk)
            acc += hs[k0 + kk] * bf2f(Wl[(k0 + kk) * 16 + o]);

        // reduce the 16 partials per output (lanes differ only in g, within half-wave)
#pragma unroll
        for (int off = 8; off; off >>= 1) acc += __shfl_xor(acc, off, 32);

        if (g == 0) {
            if (o < 11) {
                outCls[(long)row * 11 + o] = acc + bc[o];
            } else {
                const int j = o - 11;
                rbuf[j] = (acc + br[j]) * STD[j];
            }
        }
        __syncthreads();

        if (tid < 5) {
            const float b0 = boxes[(long)row * 5 + 0];
            const float b1 = boxes[(long)row * 5 + 1];
            const float b2 = boxes[(long)row * 5 + 2];
            const float b3 = boxes[(long)row * 5 + 3];
            const float b4 = boxes[(long)row * 5 + 4];
            const float bj = boxes[(long)row * 5 + tid];
            const float s = (tid < 4) ? strides[(long)row * 5 + tid] : 1.f;
            outRois[(long)row * 5 + tid] = bj * s;
            float d;
            switch (tid) {
                case 0: d = b2 * rbuf[0] + b0; break;
                case 1: d = b3 * rbuf[1] + b1; break;
                case 2: d = b2 * __expf(rbuf[2]); break;
                case 3: d = b3 * __expf(rbuf[3]); break;
                default: d = b4 * __expf(rbuf[4]); break;
            }
            outDec[(long)row * 5 + tid] = d * s;
        }
        __syncthreads();
    }
}

// ---------- launch ----------
extern "C" void kernel_launch(void* const* d_in, const int* in_sizes, int n_in,
                              void* d_out, int out_size, void* d_ws, size_t ws_size,
                              hipStream_t stream) {
    const float* features = (const float*)d_in[0];  // [2,2048,256,7,7]
    const float* boxes    = (const float*)d_in[1];  // [2,2048,5]
    const float* strides  = (const float*)d_in[2];  // [2,2048,5]
    const float* W1 = (const float*)d_in[3];        // [12544,1024]
    const float* b1 = (const float*)d_in[4];
    const float* W2 = (const float*)d_in[5];        // [1024,1024]
    const float* b2 = (const float*)d_in[6];
    const float* Wc = (const float*)d_in[7];        // [1024,11]
    const float* bc = (const float*)d_in[8];
    const float* Wr = (const float*)d_in[9];        // [1024,5]
    const float* br = (const float*)d_in[10];

    const int M = 2 * 2048;    // 4096 rows
    const int H = 1024;
    const int D = 256 * 7 * 7; // 12544

    unsigned short* Abf = (unsigned short*)d_ws;                    // [M, D] bf16 (L2-resident)
    unsigned short* W1t = Abf + (size_t)M * D;                      // [H, D] bf16
    unsigned short* W2t = W1t + (size_t)H * D;                      // [H, H] bf16
    unsigned short* h1  = W2t + (size_t)H * H;                      // [M, H] bf16
    unsigned short* h2  = h1  + (size_t)M * H;                      // [M, H] bf16

    float* outCls  = (float*)d_out;                 // [M,11]
    float* outDec  = outCls + (size_t)M * 11;       // [M,5]
    float* outRois = outDec + (size_t)M * 5;        // [M,5]

    convert_bf16<<<2048, 256, 0, stream>>>(features, Abf, (long)M * D / 4);
    transpose_bf16<<<dim3(D / 32, H / 32), dim3(32, 8), 0, stream>>>(W1, W1t, D, H);
    transpose_bf16<<<dim3(H / 32, H / 32), dim3(32, 8), 0, stream>>>(W2, W2t, H, H);

    gemm_relu_wmma<<<dim3(H / 128, M / 128), 256, 0, stream>>>(Abf, W1t, b1, h1, M, H, D);
    gemm_relu_wmma<<<dim3(H / 128, M / 128), 256, 0, stream>>>(h1, W2t, b2, h2, M, H, H);

    head_kernel<<<64, 256, 0, stream>>>(h2, Wc, bc, Wr, br, boxes, strides,
                                        outCls, outDec, outRois, M);
}
